// AttentionModel_69114613729060
// MI455X (gfx1250) — matrix operand order repaired and checked by reading, not compiled, not run on hardware
//
#include <hip/hip_runtime.h>

// ---------------------------------------------------------------------------
// MHA forward for MI455X (gfx1250, wave32, WMMA).
// All matmuls via V_WMMA_F32_16X16X32_BF16 (f32 accumulate).
// Fragment layouts per CDNA5 ISA 7.12.2:
//   A (16x32 bf16): lane m=lane&15, koff=(lane>>4)*8 -> rows, 2x16B loads
//   B (32x16 bf16): lane n=lane&15, h=lane>>4 reads B^T[n][8h..8h+7], [16+8h..16+8h+7]
//   C/D (16x16 f32): lane col n=lane&15, rows v+8*(lane>>4)
// GEMMs are double-buffered over K so loads overlap WMMA execution.
// ---------------------------------------------------------------------------

typedef __bf16 bf16;
typedef __attribute__((ext_vector_type(16))) __bf16 v16bf;
typedef __attribute__((ext_vector_type(8)))  __bf16 v8bf;
typedef __attribute__((ext_vector_type(8)))  float  v8f;

#define S_LEN   2048
#define DIM     1024
#define HEADS   16
#define HD      64
#define NTOK    8192          // B * S
#define QK_SCALE 0.125f       // 64^-0.5

union AFrag { v16bf v; struct { v8bf lo, hi; } h; };

__device__ __forceinline__ v16bf load_a_frag(const bf16* __restrict__ p, int ld) {
  // 16x32 tile, row-major source. lane -> row m, k-offset per lane half.
  const int lane = threadIdx.x & 31;
  const int m    = lane & 15;
  const int koff = (lane >> 4) << 3;
  const bf16* r = p + (size_t)m * ld + koff;
  AFrag f;
  f.h.lo = *(const v8bf*)(r);        // K = koff .. koff+7
  f.h.hi = *(const v8bf*)(r + 16);   // K = 16+koff .. 16+koff+7
  return f.v;
}

__device__ __forceinline__ v16bf load_b_frag(const bf16* __restrict__ bt, int ld) {
  // 32x16 tile given as B^T row-major (bt[n][k]). lane -> col n, k-offset per lane half.
  const int lane = threadIdx.x & 31;
  const int n    = lane & 15;
  const int h    = lane >> 4;
  const bf16* r = bt + (size_t)n * ld + (h << 3);
  AFrag f;
  f.h.lo = *(const v8bf*)(r);        // K = 8h .. 8h+7
  f.h.hi = *(const v8bf*)(r + 16);   // K = 16+8h .. 16+8h+7
  return f.v;
}

__device__ __forceinline__ v8f wmma_bf16(v16bf a, v16bf b, v8f c) {
  return __builtin_amdgcn_wmma_f32_16x16x32_bf16(false, a, false, b, (short)0, c,
                                                 false, false);
}

// ---------------------------------------------------------------------------
// fp32 -> bf16 conversion / transpose kernels
// ---------------------------------------------------------------------------
__global__ void cvt_kernel(const float* __restrict__ x, bf16* __restrict__ xb, int n) {
  int i = blockIdx.x * 256 + threadIdx.x;
  if (i < n) xb[i] = (bf16)x[i];
}

__global__ void cvt_transpose_kernel(const float* __restrict__ w,
                                     bf16* __restrict__ wt, int N) {
  // w is [1024][N] row-major; wt[n][k] = w[k][n]  (coalesced writes); K = DIM
  int i = blockIdx.x * 256 + threadIdx.x;
  int n = i >> 10, k = i & (DIM - 1);
  wt[i] = (bf16)w[(size_t)k * N + n];
}

// ---------------------------------------------------------------------------
// GEMM core: 8 waves/block, wave = (wm 0..3, wn 0..1) -> 32x64 tile,
// block tile 128x128. K loop double-buffered in steps of 64 (2 x 32).
// ---------------------------------------------------------------------------
__device__ __forceinline__ void gemm_load_buf(
    v16bf a[2], v16bf b[4],
    const bf16* __restrict__ A, const bf16* __restrict__ Bt,
    int mrow, int n0, int kk)
{
#pragma unroll
  for (int mt = 0; mt < 2; ++mt)
    a[mt] = load_a_frag(A + (size_t)(mrow + 16 * mt) * DIM + kk, DIM);
#pragma unroll
  for (int nt = 0; nt < 4; ++nt)
    b[nt] = load_b_frag(Bt + (size_t)(n0 + 16 * nt) * DIM + kk, DIM);
}

__device__ __forceinline__ void gemm_mma(v8f c[2][4], const v16bf a[2], const v16bf b[4]) {
#pragma unroll
  for (int mt = 0; mt < 2; ++mt)
#pragma unroll
    for (int nt = 0; nt < 4; ++nt)
      c[mt][nt] = wmma_bf16(a[mt], b[nt], c[mt][nt]);
}

__device__ __forceinline__ void gemm_core(
    v8f c[2][4], const bf16* __restrict__ A, const bf16* __restrict__ Bt,
    int mrow, int n0)
{
  v16bf a0[2], b0[4], a1[2], b1[4];
  gemm_load_buf(a0, b0, A, Bt, mrow, n0, 0);
  int kk = 0;
  for (; kk < DIM - 64; kk += 64) {
    gemm_load_buf(a1, b1, A, Bt, mrow, n0, kk + 32);  // in flight during...
    gemm_mma(c, a0, b0);                              // ...these 8 WMMAs
    gemm_load_buf(a0, b0, A, Bt, mrow, n0, kk + 64);
    gemm_mma(c, a1, b1);
  }
  gemm_load_buf(a1, b1, A, Bt, mrow, n0, kk + 32);
  gemm_mma(c, a0, b0);
  gemm_mma(c, a1, b1);
}

// ---------------------------------------------------------------------------
// QKV GEMM: C[8192 x 3072] = Xb * Wqkv + b, scattered to Q/K (row) and V^T.
// grid (3*DIM/128, NTOK/128)
// ---------------------------------------------------------------------------
__global__ __launch_bounds__(256) void qkv_gemm_kernel(
    const bf16* __restrict__ Xb, const bf16* __restrict__ WqkvT,
    const float* __restrict__ bias,
    bf16* __restrict__ Qb, bf16* __restrict__ Kb, bf16* __restrict__ Vt)
{
  const int wave = threadIdx.x >> 5;
  const int mrow = blockIdx.y * 128 + (wave & 3) * 32;
  const int n0   = blockIdx.x * 128 + (wave >> 2) * 64;
  const int lane = threadIdx.x & 31;

  v8f c[2][4] = {};
  gemm_core(c, Xb, WqkvT, mrow, n0);

  const int h = lane >> 4, n = lane & 15;
#pragma unroll
  for (int mt = 0; mt < 2; ++mt)
#pragma unroll
    for (int nt = 0; nt < 4; ++nt) {
      const int gn  = n0 + nt * 16 + n;
      const float bia = bias[gn];
      const int t  = gn >> 10;          // 0=Q 1=K 2=V
      const int hh = (gn >> 6) & 15;    // head
      const int d  = gn & 63;           // dim in head
#pragma unroll
      for (int v = 0; v < 8; ++v) {
        const int gm = mrow + 16 * mt + v + 8 * h;
        const int bb = gm >> 11, s = gm & 2047;
        const float val = c[mt][nt][v] + bia;
        const size_t hb = (size_t)(bb * HEADS + hh) * S_LEN * HD;
        if (t == 0)      Qb[hb + (size_t)s * HD + d] = (bf16)(val * QK_SCALE);
        else if (t == 1) Kb[hb + (size_t)s * HD + d] = (bf16)val;
        else             Vt[hb + (size_t)d * S_LEN + s] = (bf16)val;
      }
    }
}

// ---------------------------------------------------------------------------
// Attention: per wave one 16-query tile of one (b,h). Online softmax over
// 32-key tiles; scores computed transposed (K*Q^T), output as O^T = V^T*P^T.
// K fragments double-buffered across tiles; V fragments issued before the
// softmax VALU so their latency hides under it + the LDS P^T relayout.
// ---------------------------------------------------------------------------
__device__ __forceinline__ void attn_load_k(v16bf aK[4],
                                            const bf16* __restrict__ Kh, int key0) {
  aK[0] = load_a_frag(Kh + (size_t)key0 * HD, HD);
  aK[1] = load_a_frag(Kh + (size_t)key0 * HD + 32, HD);
  aK[2] = load_a_frag(Kh + (size_t)(key0 + 16) * HD, HD);
  aK[3] = load_a_frag(Kh + (size_t)(key0 + 16) * HD + 32, HD);
}

__device__ __forceinline__ void attn_tile(
    const v16bf aK[4], const bf16* __restrict__ Vh, int key0,
    v16bf bq0, v16bf bq1, bf16* myst,
    float& m_run, float& l_run, v8f o[4])
{
  const int lane = threadIdx.x & 31;
  const int n = lane & 15, h = lane >> 4;

  // scores^T tile [32 keys x 16 q] = K * Q^T  (Q pre-scaled)
  v8f p0 = {}, p1 = {};
  p0 = wmma_bf16(aK[0], bq0, p0);
  p0 = wmma_bf16(aK[1], bq1, p0);
  p1 = wmma_bf16(aK[2], bq0, p1);
  p1 = wmma_bf16(aK[3], bq1, p1);

  // issue V fragment loads now; consumed only after softmax + LDS round-trip
  v16bf aV[4];
#pragma unroll
  for (int f = 0; f < 4; ++f)
    aV[f] = load_a_frag(Vh + (size_t)(16 * f) * S_LEN + key0, S_LEN);

  // online softmax: lane owns query n; keys live in-lane (rows) + other half
  float vmax = -INFINITY;
#pragma unroll
  for (int v = 0; v < 8; ++v) {
    vmax = fmaxf(vmax, p0[v]);
    vmax = fmaxf(vmax, p1[v]);
  }
  vmax = fmaxf(vmax, __shfl_xor(vmax, 16, 32));
  const float m_new  = fmaxf(m_run, vmax);
  const float factor = __expf(m_run - m_new);
  float sum = 0.f;
  v8bf pk0, pk1;
#pragma unroll
  for (int v = 0; v < 8; ++v) {
    const float e0 = __expf(p0[v] - m_new);
    const float e1 = __expf(p1[v] - m_new);
    sum += e0 + e1;
    pk0[v] = (bf16)e0;
    pk1[v] = (bf16)e1;
  }
  sum += __shfl_xor(sum, 16, 32);
  l_run = l_run * factor + sum;
  m_run = m_new;
#pragma unroll
  for (int f = 0; f < 4; ++f)
#pragma unroll
    for (int v = 0; v < 8; ++v) o[f][v] *= factor;

  // stage P row-major [q=n][k0..31] (C-frag lane holds contiguous key rows)
  *(v8bf*)(myst + n * 32 + 8 * h)      = pk0;   // keys 8h..8h+7
  *(v8bf*)(myst + n * 32 + 16 + 8 * h) = pk1;   // keys 16+8h..
  asm volatile("s_wait_dscnt 0" ::: "memory");
  const v16bf bp = load_b_frag(myst, 32);       // B = P^T [32k x 16q]

  // O^T[f] += V^T[16d x 32k] * P^T
#pragma unroll
  for (int f = 0; f < 4; ++f) o[f] = wmma_bf16(aV[f], bp, o[f]);
}

__global__ __launch_bounds__(256) void attn_kernel(
    const bf16* __restrict__ Qb, const bf16* __restrict__ Kb,
    const bf16* __restrict__ Vt, bf16* __restrict__ AttnB)
{
  __shared__ __align__(16) bf16 stage[8][16 * 64];   // 2KB per wave
  const int wave = threadIdx.x >> 5;
  const int lane = threadIdx.x & 31;
  const int n = lane & 15, h = lane >> 4;

  const int tile = blockIdx.x * 8 + wave;      // 0..8191
  const int bh = tile >> 7;                    // (b*H+h), 128 q-tiles each
  const int q0 = (tile & 127) * 16;

  const bf16* Qh = Qb + (size_t)bh * S_LEN * HD;
  const bf16* Kh = Kb + (size_t)bh * S_LEN * HD;
  const bf16* Vh = Vt + (size_t)bh * HD * S_LEN;
  bf16* myst = stage[wave];

  // Q as B-frags (reused for every key tile): B^T row-major = Q[q][d]
  const v16bf bq0 = load_b_frag(Qh + (size_t)q0 * HD + 0, HD);   // d 0..31
  const v16bf bq1 = load_b_frag(Qh + (size_t)q0 * HD + 32, HD);  // d 32..63

  float m_run = -INFINITY, l_run = 0.f;
  v8f o[4] = {};

  v16bf aK0[4], aK1[4];
  attn_load_k(aK0, Kh, 0);
  int key0 = 0;
  for (; key0 < S_LEN - 64; key0 += 64) {
    attn_load_k(aK1, Kh, key0 + 32);
    attn_tile(aK0, Vh, key0, bq0, bq1, myst, m_run, l_run, o);
    attn_load_k(aK0, Kh, key0 + 64);
    attn_tile(aK1, Vh, key0 + 32, bq0, bq1, myst, m_run, l_run, o);
  }
  attn_load_k(aK1, Kh, key0 + 32);
  attn_tile(aK0, Vh, key0, bq0, bq1, myst, m_run, l_run, o);
  attn_tile(aK1, Vh, key0 + 32, bq0, bq1, myst, m_run, l_run, o);

  const float inv = 1.f / l_run;
  // stage O row-major [q=n][d 0..63]
#pragma unroll
  for (int f = 0; f < 4; ++f) {
    v8bf pk;
#pragma unroll
    for (int v = 0; v < 8; ++v) pk[v] = (bf16)(o[f][v] * inv);
    *(v8bf*)(myst + n * 64 + f * 16 + 8 * h) = pk;
  }
  asm volatile("s_wait_dscnt 0" ::: "memory");

  // coalesced write: lane -> (q = lane/2, 32-d half), 64B per lane
  const int q = lane >> 1, doff = (lane & 1) * 32;
  const int bb = bh >> 4, hh = bh & 15;
  bf16* dst = AttnB + (size_t)(bb * S_LEN + q0 + q) * DIM + hh * HD + doff;
  const bf16* src = myst + q * 64 + doff;
  *(v8bf*)(dst + 0)  = *(const v8bf*)(src + 0);
  *(v8bf*)(dst + 8)  = *(const v8bf*)(src + 8);
  *(v8bf*)(dst + 16) = *(const v8bf*)(src + 16);
  *(v8bf*)(dst + 24) = *(const v8bf*)(src + 24);
}

// ---------------------------------------------------------------------------
// Output projection: out[8192 x 1024] = AttnB * Wproj + b  (f32 out)
// ---------------------------------------------------------------------------
__global__ __launch_bounds__(256) void proj_gemm_kernel(
    const bf16* __restrict__ A, const bf16* __restrict__ Bt,
    const float* __restrict__ bias, float* __restrict__ out)
{
  const int wave = threadIdx.x >> 5;
  const int mrow = blockIdx.y * 128 + (wave & 3) * 32;
  const int n0   = blockIdx.x * 128 + (wave >> 2) * 64;
  const int lane = threadIdx.x & 31;

  v8f c[2][4] = {};
  gemm_core(c, A, Bt, mrow, n0);

  const int h = lane >> 4, n = lane & 15;
#pragma unroll
  for (int mt = 0; mt < 2; ++mt)
#pragma unroll
    for (int nt = 0; nt < 4; ++nt) {
      const int gn = n0 + nt * 16 + n;
      const float bia = bias[gn];
#pragma unroll
      for (int v = 0; v < 8; ++v) {
        const int gm = mrow + 16 * mt + v + 8 * h;
        out[(size_t)gm * DIM + gn] = c[mt][nt][v] + bia;
      }
    }
}

// ---------------------------------------------------------------------------
extern "C" void kernel_launch(void* const* d_in, const int* in_sizes, int n_in,
                              void* d_out, int out_size, void* d_ws, size_t ws_size,
                              hipStream_t stream) {
  (void)in_sizes; (void)n_in; (void)out_size; (void)ws_size;
  const float* x      = (const float*)d_in[0];
  const float* w_qkv  = (const float*)d_in[1];
  const float* b_qkv  = (const float*)d_in[2];
  const float* w_proj = (const float*)d_in[3];
  const float* b_proj = (const float*)d_in[4];

  char* ws = (char*)d_ws;
  size_t off = 0;
  bf16* Xb     = (bf16*)(ws + off); off += (size_t)NTOK * DIM * 2;        // 16 MB
  bf16* WqkvT  = (bf16*)(ws + off); off += (size_t)3 * DIM * DIM * 2;     //  6 MB
  bf16* WprojT = (bf16*)(ws + off); off += (size_t)DIM * DIM * 2;         //  2 MB
  bf16* Qb     = (bf16*)(ws + off); off += (size_t)NTOK * DIM * 2;        // 16 MB
  bf16* Kb     = (bf16*)(ws + off); off += (size_t)NTOK * DIM * 2;        // 16 MB
  bf16* Vt     = (bf16*)(ws + off); off += (size_t)NTOK * DIM * 2;        // 16 MB
  bf16* AttnB  = (bf16*)(ws + off); off += (size_t)NTOK * DIM * 2;        // 16 MB

  cvt_kernel<<<(NTOK * DIM) / 256, 256, 0, stream>>>(x, Xb, NTOK * DIM);
  cvt_transpose_kernel<<<(3 * DIM * DIM) / 256, 256, 0, stream>>>(w_qkv, WqkvT, 3 * DIM);
  cvt_transpose_kernel<<<(DIM * DIM) / 256, 256, 0, stream>>>(w_proj, WprojT, DIM);

  qkv_gemm_kernel<<<dim3(3 * DIM / 128, NTOK / 128), 256, 0, stream>>>(
      Xb, WqkvT, b_qkv, Qb, Kb, Vt);

  attn_kernel<<<(NTOK / 16) / 8, 256, 0, stream>>>(Qb, Kb, Vt, AttnB);

  proj_gemm_kernel<<<dim3(DIM / 128, NTOK / 128), 256, 0, stream>>>(
      AttnB, WprojT, b_proj, (float*)d_out);
}
